// PrevWindowAttentionLayer_42039139893267
// MI455X (gfx1250) — matrix operand order repaired and checked by reading, not compiled, not run on hardware
//
#include <hip/hip_runtime.h>
#include <hip/hip_bf16.h>

// ---------------------------------------------------------------------------
// Problem constants (from reference): B=8, T=32, N=500, Dp=200, Dc=128,
// HEADS=4, UNITS=64, MSG_UNITS=64.  N padded to 512 for WMMA tiling.
// ---------------------------------------------------------------------------
#define BQ 8
#define TQ 32
#define NQ 500
#define NPAD 512
#define DPREV 200
#define DCURR 128
#define HEADS 4
#define UNITS 64
#define MU 64
#define KCH 128     // k-chunk staged into LDS per pipeline stage in k4

typedef __attribute__((ext_vector_type(16))) _Float16 v16h;
typedef __attribute__((ext_vector_type(8)))  _Float16 v8h;
typedef __attribute__((ext_vector_type(8)))  float    v8f;
typedef __attribute__((ext_vector_type(4)))  float    v4f;

static __device__ __forceinline__ v16h cat16(v8h lo, v8h hi) {
    return __builtin_shufflevector(lo, hi, 0, 1, 2, 3, 4, 5, 6, 7,
                                           8, 9, 10, 11, 12, 13, 14, 15);
}

static __device__ __forceinline__ v8f wmma_f16(v16h a, v16h b, v8f c) {
    // D(f32) = A(f16,16x32) * B(f16,32x16) + C(f32)
    return __builtin_amdgcn_wmma_f32_16x16x32_f16(
        /*neg_a=*/false, a, /*neg_b=*/false, b,
        /*c_mod=*/(short)0, c, /*reuse_a=*/false, /*reuse_b=*/false);
}

// A-operand (16x32 f16): lane<16 -> row n0+lane, K = {k0..k0+7, k0+16..k0+23}
//                        lane>=16 -> row n0+lane-16, K = {k0+8..+15, k0+24..+31}
static __device__ __forceinline__ v16h load_A_f16(const _Float16* rowPtr, int k0, int lane) {
    int koff = k0 + ((lane & 16) ? 8 : 0);
    v8h lo = *(const v8h*)(rowPtr + koff);
    v8h hi = *(const v8h*)(rowPtr + koff + 16);
    return cat16(lo, hi);
}

// B-operand (32x16 f16): lane holds column u0+(lane&15), K = {k0+8h..+7, k0+16+8h..+7}
// with h = lane>>4  (colPtr points at column start, K contiguous in memory).
static __device__ __forceinline__ v16h load_B_f16(const _Float16* colPtr, int k0, int lane) {
    int koff = k0 + ((lane & 16) ? 8 : 0);
    v8h lo = *(const v8h*)(colPtr + koff);
    v8h hi = *(const v8h*)(colPtr + koff + 16);
    return cat16(lo, hi);
}

// ---------------------------------------------------------------------------
// Kernel 1: h_src / h_dst projections, FP32 (accuracy feeds softmax).
// grid = HEADS*B*N blocks of 128 threads. threads 0..63 -> src, 64..127 -> dst.
// ---------------------------------------------------------------------------
__global__ void k1_proj_srcdst(const float* __restrict__ prev,
                               const float* __restrict__ Wsrc,
                               const float* __restrict__ Wdst,
                               float* __restrict__ hsrc,
                               float* __restrict__ hdst) {
    int blk = blockIdx.x;
    int h = blk / (BQ * NQ);
    int rem = blk % (BQ * NQ);
    int b = rem / NQ;
    int n = rem % NQ;
    int t = threadIdx.x;

    __shared__ float prow[DPREV];
    for (int d = t; d < DPREV; d += 128)
        prow[d] = prev[((long)b * NQ + n) * DPREV + d];
    __syncthreads();

    int u = t & 63;
    const float* W = (t & 64) ? Wdst : Wsrc;
    float* dst = (t & 64) ? hdst : hsrc;
    const float* wcol = W + (long)h * DPREV * UNITS + u;

    float acc = 0.f;
#pragma unroll 4
    for (int d = 0; d < DPREV; ++d)
        acc = fmaf(prow[d], wcol[(long)d * UNITS], acc);

    dst[((((long)h * BQ + b) * NQ) + n) * UNITS + u] = acc;
}

// ---------------------------------------------------------------------------
// Kernel 2: GATv2 scores + softmax -> attn (f16, padded [H][B][512][512]).
// grid = HEADS*B*512 blocks of 256 threads; each block = one attention row i.
// ---------------------------------------------------------------------------
__global__ void k2_scores_softmax(const float* __restrict__ hsrc,
                                  const float* __restrict__ hdst,
                                  const float* __restrict__ avec,
                                  _Float16* __restrict__ attn) {
    int blk = blockIdx.x;
    int h = blk >> 12;          // / (B*512) = /4096
    int rem = blk & 4095;
    int b = rem >> 9;
    int i = rem & 511;
    int t = threadIdx.x;

    _Float16* row = attn + ((((long)h * BQ + b) * NPAD + i) * NPAD);
    if (i >= NQ) {              // padded rows: all-zero
        row[t] = (_Float16)0.f;
        row[t + 256] = (_Float16)0.f;
        return;
    }

    __shared__ float ssrc[UNITS];
    __shared__ float sa[UNITS];
    __shared__ float red[256];
    if (t < UNITS)
        ssrc[t] = hsrc[((((long)h * BQ + b) * NQ) + i) * UNITS + t];
    else if (t < 2 * UNITS)
        sa[t - UNITS] = avec[h * UNITS + (t - UNITS)];
    __syncthreads();

    const float* hdst_hb = hdst + (((long)h * BQ + b) * NQ) * UNITS;

    // Each thread scores columns j = t and j = t + 256.
    float s[2];
#pragma unroll
    for (int c = 0; c < 2; ++c) {
        int j = t + c * 256;
        if (j < NQ) {
            const v4f* dj = (const v4f*)(hdst_hb + (long)j * UNITS);
            float acc = 0.f;
#pragma unroll 4
            for (int k = 0; k < UNITS / 4; ++k) {
                v4f d4 = dj[k];
#pragma unroll
                for (int e = 0; e < 4; ++e) {
                    float x = ssrc[4 * k + e] + d4[e];
                    x = (x > 0.f) ? x : 0.2f * x;   // leaky_relu 0.2
                    acc = fmaf(x, sa[4 * k + e], acc);
                }
            }
            s[c] = acc;
        } else {
            s[c] = -3.0e38f;    // padded column -> exp()=0
        }
    }

    // row max
    red[t] = fmaxf(s[0], s[1]);
    __syncthreads();
    for (int w = 128; w > 0; w >>= 1) {
        if (t < w) red[t] = fmaxf(red[t], red[t + w]);
        __syncthreads();
    }
    float m = red[0];
    __syncthreads();

    float e0 = __expf(s[0] - m);
    float e1 = (t + 256 < NQ) ? __expf(s[1] - m) : 0.f;

    // row sum
    red[t] = e0 + e1;
    __syncthreads();
    for (int w = 128; w > 0; w >>= 1) {
        if (t < w) red[t] += red[t + w];
        __syncthreads();
    }
    float inv = 1.f / red[0];

    row[t] = (_Float16)(e0 * inv);
    row[t + 256] = (_Float16)(e1 * inv);
}

// ---------------------------------------------------------------------------
// Kernel 3: msg = curr_hidden @ W_msg  (WMMA f16, f32 acc), result stored
// TRANSPOSED as msgT[h][b][t][u][m] (m padded to 512, zeros for m>=500).
// grid = HEADS*B*T blocks of 256 threads (8 waves); W_msg^T staged in LDS.
// Outer loops at unroll 1 (bounded pressure); B tiles loaded as a group so
// ds_loads clause up and overlap with WMMA instead of lockstep dscnt-0 waits.
// ---------------------------------------------------------------------------
__global__ void k3_msg_proj(const float* __restrict__ curr,
                            const float* __restrict__ Wmsg,
                            _Float16* __restrict__ msgT) {
    int blk = blockIdx.x;           // H*B*T = 1024
    int h = blk >> 8;
    int rem = blk & 255;
    int b = rem >> 5;
    int t = rem & 31;
    int tid = threadIdx.x;

    // W_msg^T in LDS: [u][d] f16, 64*128 = 16 KB
    __shared__ __align__(16) _Float16 wlds[MU * DCURR];
    for (int idx = tid; idx < MU * DCURR; idx += 256) {
        int u = idx >> 7, d = idx & 127;
        wlds[idx] = (_Float16)Wmsg[((long)h * DCURR + d) * MU + u];
    }
    __syncthreads();

    int wave = tid >> 5, lane = tid & 31;
    int lane15 = lane & 15, hiH = lane >> 4;

    const float* Abase = curr + (((long)b * TQ + t) * NQ) * DCURR;
    _Float16* obase = msgT + ((((long)h * BQ + b) * TQ + t) * MU) * NPAD;

#pragma unroll 1
    for (int rt = wave; rt < NPAD / 16; rt += 8) {   // 4 row-tiles per wave
        int n0 = rt * 16;
        int rowm = n0 + lane15;
        v8f acc[4] = {v8f{}, v8f{}, v8f{}, v8f{}};
        const float* Arow = Abase + (long)rowm * DCURR;

#pragma unroll 1
        for (int kk = 0; kk < DCURR / 32; ++kk) {
            int k0 = kk * 32;
            v16h A = {};
            if (rowm < NQ) {                 // zero A-tile => zero padding rows
                const float* p = Arow + k0 + (hiH ? 8 : 0);
#pragma unroll
                for (int e = 0; e < 8; ++e) {
                    A[e]     = (_Float16)p[e];
                    A[8 + e] = (_Float16)p[16 + e];
                }
            }
            // Load all 4 B tiles first (independent regs), then issue WMMAs.
            v16h Bv[4];
#pragma unroll
            for (int ut = 0; ut < 4; ++ut)
                Bv[ut] = load_B_f16(wlds + (long)(ut * 16 + lane15) * DCURR, k0, lane);
#pragma unroll
            for (int ut = 0; ut < 4; ++ut)
                acc[ut] = wmma_f16(A, Bv[ut], acc[ut]);
        }
        // D lane L: col u = ut*16 + (L&15), rows r + 8*(L>=16) -> 8 contiguous m
#pragma unroll
        for (int ut = 0; ut < 4; ++ut) {
            v8h st = __builtin_convertvector(acc[ut], v8h);
            *(v8h*)(obase + (long)(ut * 16 + lane15) * NPAD + n0 + hiH * 8) = st;
        }
    }
}

// ---------------------------------------------------------------------------
// Kernel 4: out[b,t,n,u] = (1/4) * sum_h attn_h[n,:] @ msgT_h
// grid = B*T*4 blocks of 256 threads (8 waves).  B-panel (msgT slice) is
// streamed into LDS with gfx1250 async global->LDS DMA, double-buffered:
// 16 stages of (head h, k-chunk c of 128); each wave issues 4x
// global_load_async_to_lds_b128 (512B each -> 16KB/block/stage), waits on
// ASYNCcnt, barriers, then runs 16 WMMAs out of LDS.
// ---------------------------------------------------------------------------
__global__ void k4_attn_apply(const _Float16* __restrict__ attn,
                              const _Float16* __restrict__ msgT,
                              float* __restrict__ out) {
    int blk = blockIdx.x;       // B*T*4 = 1024
    int bt = blk >> 2;
    int mblk = blk & 3;
    int b = bt >> 5;
    int t = bt & 31;
    int tid = threadIdx.x;
    int wave = tid >> 5, lane = tid & 31;
    int lane15 = lane & 15, hiH = lane >> 4;

    int n0 = (mblk * 8 + wave) * 16;

    // double-buffered B panel: [u 0..63][k-local 0..127] f16 = 16 KB x 2
    __shared__ __align__(16) _Float16 sb[2][MU * KCH];
    // flat LDS pointer low 32 bits == wave-relative LDS byte address (ISA 10.2)
    unsigned sbOff[2];
    sbOff[0] = (unsigned)(unsigned long long)(&sb[0][0]);
    sbOff[1] = (unsigned)(unsigned long long)(&sb[1][0]);

    v8f acc[4] = {v8f{}, v8f{}, v8f{}, v8f{}};

    // ---- async stage issue: stage s = (head h = s>>2, k-chunk c = s&3) ----
    auto issue_stage = [&](int s) {
        int h = s >> 2, c = s & 3;
        const _Float16* gbase =
            msgT + ((((long)h * BQ + b) * TQ + t) * MU) * NPAD;   // uniform
        unsigned sbo = sbOff[s & 1];
#pragma unroll
        for (int j = 0; j < 4; ++j) {
            int u = wave * 8 + j * 2 + hiH;                 // 2 u-rows / instr
            unsigned voff  = (unsigned)((u * NPAD + c * KCH + lane15 * 8) * 2);
            unsigned laddr = sbo + (unsigned)((u * KCH + lane15 * 8) * 2);
            asm volatile("global_load_async_to_lds_b128 %0, %1, %2"
                         :: "v"(laddr), "v"(voff), "s"(gbase)
                         : "memory");
        }
    };

    issue_stage(0);

#pragma unroll 1
    for (int s = 0; s < 16; ++s) {
        if (s < 15) {
            issue_stage(s + 1);                       // prefetch next chunk
            asm volatile("s_wait_asynccnt 0x4" ::: "memory"); // stage s landed
        } else {
            asm volatile("s_wait_asynccnt 0x0" ::: "memory");
        }
        __syncthreads();                              // all waves' DMA visible

        int h = s >> 2, c = s & 3;
        const _Float16* Arow =
            attn + ((((long)h * BQ + b) * NPAD) + n0 + lane15) * NPAD + c * KCH;
        const _Float16* sbp = &sb[s & 1][0];

#pragma unroll
        for (int kk = 0; kk < KCH / 32; ++kk) {
            int k0 = kk * 32;
            v16h A = load_A_f16(Arow, k0, lane);
            // Load all 4 B tiles first (independent regs), then issue WMMAs.
            v16h Bv[4];
#pragma unroll
            for (int ut = 0; ut < 4; ++ut)
                Bv[ut] = load_B_f16(sbp + (ut * 16 + lane15) * KCH, k0, lane);
#pragma unroll
            for (int ut = 0; ut < 4; ++ut)
                acc[ut] = wmma_f16(A, Bv[ut], acc[ut]);
        }
        __syncthreads();        // buffer s&1 free before stage s+2 overwrites
    }

    // Mean over heads (x0.25) and store fp32 with n<500 guard.
    float* obase = out + (((long)b * TQ + t) * NQ) * MU;
#pragma unroll
    for (int ut = 0; ut < 4; ++ut) {
        int u = ut * 16 + lane15;
#pragma unroll
        for (int r = 0; r < 8; ++r) {
            int n = n0 + hiH * 8 + r;
            if (n < NQ)
                obase[(long)n * MU + u] = acc[ut][r] * 0.25f;
        }
    }
}

// ---------------------------------------------------------------------------
// Host-side launcher.
// Inputs (setup_inputs order): prev_features[B,N,200] f32, curr_hidden
// [B,T,N,128] f32, W_src[4,200,64], W_dst[4,200,64], a[4,64,1], W_msg[4,128,64].
// Output: [B,T,N,64] f32.
// Workspace layout (bytes):
//   hsrc  @ 0          : H*B*N*64 f32  = 4,096,000
//   hdst  @ 4,096,000  : H*B*N*64 f32  = 4,096,000
//   attn  @ 8,192,000  : H*B*512*512 f16 = 16,777,216
//   msgT  @ 16,580,608 : H*B*T*64*512 f16 = 67,108,864   (total ~83.7 MB)
// ---------------------------------------------------------------------------
extern "C" void kernel_launch(void* const* d_in, const int* in_sizes, int n_in,
                              void* d_out, int out_size, void* d_ws, size_t ws_size,
                              hipStream_t stream) {
    (void)in_sizes; (void)n_in; (void)out_size; (void)ws_size;

    const float* prev = (const float*)d_in[0];
    const float* curr = (const float*)d_in[1];
    const float* Wsrc = (const float*)d_in[2];
    const float* Wdst = (const float*)d_in[3];
    const float* avec = (const float*)d_in[4];
    const float* Wmsg = (const float*)d_in[5];
    float* out = (float*)d_out;

    char* ws = (char*)d_ws;
    float*     hsrc = (float*)(ws);
    float*     hdst = (float*)(ws + 4096000);
    _Float16*  attn = (_Float16*)(ws + 8192000);
    _Float16*  msgT = (_Float16*)(ws + 16580608);

    k1_proj_srcdst<<<HEADS * BQ * NQ, 128, 0, stream>>>(prev, Wsrc, Wdst, hsrc, hdst);
    k2_scores_softmax<<<HEADS * BQ * NPAD, 256, 0, stream>>>(hsrc, hdst, avec, attn);
    k3_msg_proj<<<HEADS * BQ * TQ, 256, 0, stream>>>(curr, Wmsg, msgT);
    k4_attn_apply<<<BQ * TQ * 4, 256, 0, stream>>>(attn, msgT, out);
}
